// IntraBandBiMamba_16844861735721
// MI455X (gfx1250) — compile-verified, matrix operand order checked by reading, not run on hardware
//
#include <hip/hip_runtime.h>
#include <hip/hip_bf16.h>
#include <math.h>

// ---------------------------------------------------------------------------
// IntraBandBiMamba for MI455X (gfx1250): all GEMMs via v_wmma_f32_16x16x32_bf16
// Shapes: B=4, T=256, K=30, D_MODEL=D_INNER=128, D_STATE=16, DT_RANK=8, D_CONV=4
// M = B*K*T = 30720 rows for every GEMM.
// GEMM waves are register-blocked 4 M-tiles x 1 N-tile: the B (weight)
// fragment is loaded once per K-step and reused by 4 independent WMMAs,
// which also breaks the WMMA->WMMA D-register RAW hazard chain.
// ---------------------------------------------------------------------------

typedef __bf16 bf16;
typedef __attribute__((ext_vector_type(16))) __bf16 v16bf;
typedef __attribute__((ext_vector_type(8)))  __bf16 bf16x8;
typedef __attribute__((ext_vector_type(8)))  float  v8f;

#define WMMA_BF16(a, b, c) \
  __builtin_amdgcn_wmma_f32_16x16x32_bf16(false, (a), false, (b), (short)0, (c), false, false)

// Load a 16-element bf16 fragment for one lane: row base pointer `p`,
// K-offset k0 (multiple of 32). ISA layout: lanes 0-15 take K {0..7,16..23},
// lanes 16-31 take K {8..15,24..31}. Two 16-byte vector loads.
__device__ __forceinline__ v16bf load_frag_row(const bf16* p, int k0, int lane) {
  int hi = (lane >> 4) & 1;
  bf16x8 lo8 = *reinterpret_cast<const bf16x8*>(p + k0 + hi * 8);
  bf16x8 hi8 = *reinterpret_cast<const bf16x8*>(p + k0 + hi * 8 + 16);
  v16bf r;
#pragma unroll
  for (int i = 0; i < 8; ++i) { r[i] = lo8[i]; r[i + 8] = hi8[i]; }
  return r;
}

// ---------------------------------------------------------------------------
// Weight conversion helpers
// ---------------------------------------------------------------------------
__global__ void cvt_f32_to_bf16(const float* __restrict__ s, bf16* __restrict__ d, int n) {
  int i = blockIdx.x * 256 + threadIdx.x;
  if (i < n) d[i] = (bf16)s[i];
}

// xproj_w is (40,128); pad to (48,128) with zero rows so N is a multiple of 16.
__global__ void cvt_pad_xproj(const float* __restrict__ s, bf16* __restrict__ d) {
  int i = blockIdx.x * 256 + threadIdx.x;
  if (i < 48 * 128) {
    int row = i >> 7;
    d[i] = (row < 40) ? (bf16)s[i] : (bf16)0.0f;
  }
}

// ---------------------------------------------------------------------------
// LayerNorm: x (B,128,T,K) f32 -> xn_h (B*K, T, 128) bf16
// One block per (b,t): tile = 128x30 slab in LDS, reduce over N per k.
// ---------------------------------------------------------------------------
__global__ void ln_kernel(const float* __restrict__ x, const float* __restrict__ lw,
                          const float* __restrict__ lb, bf16* __restrict__ xnh) {
  __shared__ float tile[128 * 30];
  __shared__ float smu[30];
  __shared__ float srs[30];
  int bt = blockIdx.x;
  int b = bt >> 8, t = bt & 255;
  int tid = threadIdx.x;
  const float* xb = x + (((size_t)b * 128) * 256 + t) * 30;
  for (int e = tid; e < 3840; e += 256) {
    int n = e / 30, k = e - n * 30;
    tile[e] = xb[(size_t)n * 256 * 30 + k];
  }
  __syncthreads();
  if (tid < 30) {
    float s = 0.f, ss = 0.f;
#pragma unroll 4
    for (int n = 0; n < 128; ++n) {
      float v = tile[n * 30 + tid];
      s += v; ss += v * v;
    }
    float mu = s * (1.0f / 128.0f);
    float var = ss * (1.0f / 128.0f) - mu * mu;
    smu[tid] = mu;
    srs[tid] = rsqrtf(var + 1e-5f);
  }
  __syncthreads();
  for (int e = tid; e < 3840; e += 256) {
    int k = e >> 7, n = e & 127;  // n-fast => contiguous bf16 writes
    float v = (tile[n * 30 + k] - smu[k]) * srs[k] * lw[n] + lb[n];
    xnh[(((size_t)(b * 30 + k)) * 256 + t) * 128 + n] = (bf16)v;
  }
}

// ---------------------------------------------------------------------------
// Generic bf16 WMMA GEMM: C[M x N] = A[M x K] * W[N x K]^T
// One wave per 64x16 output block (4 M-tiles, B fragment reused 4x).
// grid = (M/64, N/16), block = 32.
// flipA: reverse the time index (row % 256) of A reads (backward direction).
// storeBf16: write C as bf16 instead of f32.
// ---------------------------------------------------------------------------
__global__ void gemm_bf16_wmma(const bf16* __restrict__ A, int lda,
                               const bf16* __restrict__ W, int ldw,
                               void* __restrict__ Cout, int ldc,
                               int K, int flipA, int storeBf16) {
  int lane = threadIdx.x;  // block = 32 (one wave32)
  int tileM0 = blockIdx.x * 4, tileN = blockIdx.y;
  const bf16* Arow[4];
#pragma unroll
  for (int i = 0; i < 4; ++i) {
    int mrow = (tileM0 + i) * 16 + (lane & 15);
    if (flipA) { mrow = (mrow & ~255) + (255 - (mrow & 255)); }
    Arow[i] = A + (size_t)mrow * lda;
  }
  int ncol = tileN * 16 + (lane & 15);
  const bf16* Wrow = W + (size_t)ncol * ldw;
  v8f c0 = {}, c1 = {}, c2 = {}, c3 = {};
  for (int k0 = 0; k0 < K; k0 += 32) {
    v16bf b = load_frag_row(Wrow, k0, lane);
    v16bf a0 = load_frag_row(Arow[0], k0, lane);
    v16bf a1 = load_frag_row(Arow[1], k0, lane);
    v16bf a2 = load_frag_row(Arow[2], k0, lane);
    v16bf a3 = load_frag_row(Arow[3], k0, lane);
    c0 = WMMA_BF16(a0, b, c0);
    c1 = WMMA_BF16(a1, b, c1);
    c2 = WMMA_BF16(a2, b, c2);
    c3 = WMMA_BF16(a3, b, c3);
  }
  int hi = (lane >> 4) & 1;
  int n = tileN * 16 + (lane & 15);
  v8f acc[4] = {c0, c1, c2, c3};
  if (storeBf16) {
    bf16* C = (bf16*)Cout;
#pragma unroll
    for (int i = 0; i < 4; ++i)
#pragma unroll
      for (int r = 0; r < 8; ++r) {
        int m = (tileM0 + i) * 16 + r + hi * 8;
        C[(size_t)m * ldc + n] = (bf16)acc[i][r];
      }
  } else {
    float* C = (float*)Cout;
#pragma unroll
    for (int i = 0; i < 4; ++i)
#pragma unroll
      for (int r = 0; r < 8; ++r) {
        int m = (tileM0 + i) * 16 + r + hi * 8;
        C[(size_t)m * ldc + n] = acc[i][r];
      }
  }
}

// ---------------------------------------------------------------------------
// Depthwise causal conv (width 4) + SiLU on the xi half of xz.
// Block = 128 (one channel/thread), grid = 120 (one sequence/block), per dir.
// Reads xz cols [0,128); writes bf16 xi_h.
// ---------------------------------------------------------------------------
__global__ void conv_silu_kernel(const float* __restrict__ xz, const float* __restrict__ cw,
                                 const float* __restrict__ cb, bf16* __restrict__ xih) {
  int seq = blockIdx.x, ch = threadIdx.x;
  float w0 = cw[ch * 4 + 0], w1 = cw[ch * 4 + 1], w2 = cw[ch * 4 + 2], w3 = cw[ch * 4 + 3];
  float bias = cb[ch];
  float x3 = 0.f, x2 = 0.f, x1 = 0.f;
  const float* src = xz + (size_t)seq * 256 * 256 + ch;
  bf16* dst = xih + (size_t)seq * 256 * 128 + ch;
  for (int t = 0; t < 256; ++t) {
    float x0 = src[(size_t)t * 256];
    float acc = w0 * x3 + w1 * x2 + w2 * x1 + w3 * x0 + bias;
    float s = acc / (1.0f + __expf(-acc));  // SiLU
    dst[(size_t)t * 128] = (bf16)s;
    x3 = x2; x2 = x1; x1 = x0;
  }
}

// ---------------------------------------------------------------------------
// Selective scan. Block = 128 channels, grid = 120 sequences, per dir.
// Per t: dt = softplus(dbc[:8] . dtw + b); h[s] = exp(dt*A)*h + dt*B[s]*x;
// y = sum h*C; out = (y + x*D) * silu(z). Writes bf16 y for out-proj GEMM.
// ---------------------------------------------------------------------------
__global__ void scan_kernel(const float* __restrict__ dbc, const bf16* __restrict__ xih,
                            const float* __restrict__ xz, const float* __restrict__ dtw,
                            const float* __restrict__ dtb, const float* __restrict__ Alog,
                            const float* __restrict__ Dp, bf16* __restrict__ yh) {
  __shared__ float sd[48];
  int seq = blockIdx.x, ch = threadIdx.x;
  float w[8];
#pragma unroll
  for (int r = 0; r < 8; ++r) w[r] = dtw[ch * 8 + r];
  float bias = dtb[ch];
  float nA[16];
#pragma unroll
  for (int s = 0; s < 16; ++s) nA[s] = -__expf(Alog[ch * 16 + s]);  // A = -exp(A_log)
  float Dv = Dp[ch];
  float h[16];
#pragma unroll
  for (int s = 0; s < 16; ++s) h[s] = 0.f;
  const float* drow = dbc + (size_t)seq * 256 * 48;
  const bf16* xrow = xih + (size_t)seq * 256 * 128 + ch;
  const float* zrow = xz + (size_t)seq * 256 * 256 + 128 + ch;  // z half
  bf16* yrow = yh + (size_t)seq * 256 * 128 + ch;
  for (int t = 0; t < 256; ++t) {
    __syncthreads();
    if (ch < 48) sd[ch] = drow[t * 48 + ch];
    __syncthreads();
    float dtr = bias;
#pragma unroll
    for (int r = 0; r < 8; ++r) dtr += w[r] * sd[r];
    float dt = (dtr > 20.f) ? dtr : log1pf(__expf(dtr));  // softplus
    float xt = (float)xrow[(size_t)t * 128];
    float zt = zrow[(size_t)t * 256];
    float y = 0.f;
#pragma unroll
    for (int s = 0; s < 16; ++s) {
      float dA = __expf(dt * nA[s]);
      h[s] = dA * h[s] + dt * sd[8 + s] * xt;   // B = dbc[8..24)
      y += h[s] * sd[24 + s];                    // C = dbc[24..40)
    }
    float g = zt / (1.0f + __expf(-zt));         // silu(z)
    yrow[(size_t)t * 128] = (bf16)((y + xt * Dv) * g);
  }
}

// ---------------------------------------------------------------------------
// Final: out = [out_f | flip(out_b)] @ c_w^T + c_b + residual, scattered to
// d_out in (B, N, T, K) layout. One wave per 64x16 block, K = 256 (2 halves),
// B fragment reused across the 4 M-tiles.
// ---------------------------------------------------------------------------
__global__ void final_gemm(const bf16* __restrict__ of, const bf16* __restrict__ ob,
                           const bf16* __restrict__ wc, const float* __restrict__ cb,
                           const float* __restrict__ x, float* __restrict__ out) {
  int lane = threadIdx.x;
  int tileM0 = blockIdx.x * 4, tileN = blockIdx.y;
  const bf16* Af[4];
  const bf16* Ab[4];
#pragma unroll
  for (int i = 0; i < 4; ++i) {
    int mrow = (tileM0 + i) * 16 + (lane & 15);
    int seq = mrow >> 8, t = mrow & 255;
    Af[i] = of + (size_t)mrow * 128;
    Ab[i] = ob + ((size_t)(seq * 256 + (255 - t))) * 128;  // un-flip bwd dir
  }
  int ncol = tileN * 16 + (lane & 15);
  const bf16* Wrow = wc + (size_t)ncol * 256;
  v8f c0 = {}, c1 = {}, c2 = {}, c3 = {};
  for (int k0 = 0; k0 < 128; k0 += 32) {
    v16bf b = load_frag_row(Wrow, k0, lane);
    c0 = WMMA_BF16(load_frag_row(Af[0], k0, lane), b, c0);
    c1 = WMMA_BF16(load_frag_row(Af[1], k0, lane), b, c1);
    c2 = WMMA_BF16(load_frag_row(Af[2], k0, lane), b, c2);
    c3 = WMMA_BF16(load_frag_row(Af[3], k0, lane), b, c3);
  }
  for (int k0 = 0; k0 < 128; k0 += 32) {
    v16bf b = load_frag_row(Wrow, k0 + 128, lane);
    c0 = WMMA_BF16(load_frag_row(Ab[0], k0, lane), b, c0);
    c1 = WMMA_BF16(load_frag_row(Ab[1], k0, lane), b, c1);
    c2 = WMMA_BF16(load_frag_row(Ab[2], k0, lane), b, c2);
    c3 = WMMA_BF16(load_frag_row(Ab[3], k0, lane), b, c3);
  }
  int hi = (lane >> 4) & 1;
  int n = tileN * 16 + (lane & 15);
  v8f acc[4] = {c0, c1, c2, c3};
#pragma unroll
  for (int i = 0; i < 4; ++i)
#pragma unroll
    for (int r = 0; r < 8; ++r) {
      int m = (tileM0 + i) * 16 + r + hi * 8;
      int sm = m >> 8, tm = m & 255;
      int bb = sm / 30, kk = sm - bb * 30;
      size_t idx = (((size_t)bb * 128 + n) * 256 + tm) * 30 + kk;
      out[idx] = acc[i][r] + cb[n] + x[idx];  // residual = xr at same flat index
    }
}

// ---------------------------------------------------------------------------
// Host side
// ---------------------------------------------------------------------------
extern "C" void kernel_launch(void* const* d_in, const int* in_sizes, int n_in,
                              void* d_out, int out_size, void* d_ws, size_t ws_size,
                              hipStream_t stream) {
  (void)in_sizes; (void)n_in; (void)out_size; (void)ws_size;
  const size_t M = 30720;  // B*K*T

  const float* x    = (const float*)d_in[0];
  const float* ln_w = (const float*)d_in[1];
  const float* ln_b = (const float*)d_in[2];
  // per-direction params: [dir][0..8] = in_w, conv_w, conv_b, xproj_w,
  // dtproj_w, dtproj_b, A_log, D, out_w
  const float* P[2][9];
  for (int d = 0; d < 2; ++d)
    for (int i = 0; i < 9; ++i)
      P[d][i] = (const float*)d_in[3 + d * 9 + i];
  const float* c_w = (const float*)d_in[21];
  const float* c_b = (const float*)d_in[22];
  float* out = (float*)d_out;

  // Workspace carve-up (256B aligned)
  char* base = (char*)d_ws;
  size_t off = 0;
  auto carve = [&](size_t bytes) -> char* {
    off = (off + 255) & ~(size_t)255;
    char* p = base + off;
    off += bytes;
    return p;
  };
  bf16*  xnh  = (bf16*)carve(M * 128 * sizeof(bf16));            // LN output
  float* xz0  = (float*)carve(M * 256 * sizeof(float));          // fwd xz
  float* xz1  = (float*)carve(M * 256 * sizeof(float));          // bwd xz
  bf16*  xih0 = (bf16*)carve(M * 128 * sizeof(bf16));            // conv+silu out
  bf16*  xih1 = (bf16*)carve(M * 128 * sizeof(bf16));
  float* dbc0 = (float*)carve(M * 48 * sizeof(float));           // x-proj out (padded)
  float* dbc1 = (float*)carve(M * 48 * sizeof(float));
  bf16*  yh0  = (bf16*)carve(M * 128 * sizeof(bf16));            // scan out
  bf16*  yh1  = (bf16*)carve(M * 128 * sizeof(bf16));
  bf16*  oh0  = (bf16*)carve(M * 128 * sizeof(bf16));            // out-proj out
  bf16*  oh1  = (bf16*)carve(M * 128 * sizeof(bf16));
  bf16*  win[2]  = {(bf16*)carve(256 * 128 * sizeof(bf16)), (bf16*)carve(256 * 128 * sizeof(bf16))};
  bf16*  wxp[2]  = {(bf16*)carve(48 * 128 * sizeof(bf16)),  (bf16*)carve(48 * 128 * sizeof(bf16))};
  bf16*  wout[2] = {(bf16*)carve(128 * 128 * sizeof(bf16)), (bf16*)carve(128 * 128 * sizeof(bf16))};
  bf16*  wcb  = (bf16*)carve(128 * 256 * sizeof(bf16));

  float* xz[2]  = {xz0, xz1};
  bf16*  xih[2] = {xih0, xih1};
  float* dbc[2] = {dbc0, dbc1};
  bf16*  yh[2]  = {yh0, yh1};
  bf16*  oh[2]  = {oh0, oh1};

  // 1) Weight conversions (f32 -> bf16, pad x-proj to 48 rows)
  for (int d = 0; d < 2; ++d) {
    cvt_f32_to_bf16<<<(256 * 128 + 255) / 256, 256, 0, stream>>>(P[d][0], win[d], 256 * 128);
    cvt_pad_xproj  <<<(48 * 128 + 255) / 256, 256, 0, stream>>>(P[d][3], wxp[d]);
    cvt_f32_to_bf16<<<(128 * 128 + 255) / 256, 256, 0, stream>>>(P[d][8], wout[d], 128 * 128);
  }
  cvt_f32_to_bf16<<<(128 * 256 + 255) / 256, 256, 0, stream>>>(c_w, wcb, 128 * 256);

  // 2) LayerNorm -> bf16 xn
  ln_kernel<<<1024, 256, 0, stream>>>(x, ln_w, ln_b, xnh);

  // 3) In-projection GEMM: xz[dir] = (maybe time-flipped) xn @ in_w^T  (N=256)
  for (int d = 0; d < 2; ++d)
    gemm_bf16_wmma<<<dim3(M / 64, 16), 32, 0, stream>>>(
        xnh, 128, win[d], 128, xz[d], 256, 128, /*flipA=*/d, /*bf16out=*/0);

  // 4) Depthwise causal conv + SiLU on xi half
  for (int d = 0; d < 2; ++d)
    conv_silu_kernel<<<120, 128, 0, stream>>>(xz[d], P[d][1], P[d][2], xih[d]);

  // 5) x-projection GEMM: dbc = xi @ xproj_w^T  (N padded 40->48)
  for (int d = 0; d < 2; ++d)
    gemm_bf16_wmma<<<dim3(M / 64, 3), 32, 0, stream>>>(
        xih[d], 128, wxp[d], 128, dbc[d], 48, 128, 0, 0);

  // 6) Selective scan (dt-proj + softplus fused in)
  for (int d = 0; d < 2; ++d)
    scan_kernel<<<120, 128, 0, stream>>>(dbc[d], xih[d], xz[d],
                                         P[d][4], P[d][5], P[d][6], P[d][7], yh[d]);

  // 7) Out-projection GEMM: oh = y @ out_w^T (bf16 output feeds final GEMM)
  for (int d = 0; d < 2; ++d)
    gemm_bf16_wmma<<<dim3(M / 64, 8), 32, 0, stream>>>(
        yh[d], 128, wout[d], 128, oh[d], 128, 128, 0, /*bf16out=*/1);

  // 8) Final: concat (with un-flip of bwd) @ c_w^T + c_b + residual -> d_out
  final_gemm<<<dim3(M / 64, 8), 32, 0, stream>>>(oh[0], oh[1], wcb, c_b, x, out);
}
